// Linear_3728031613116
// MI455X (gfx1250) — matrix operand order repaired and checked by reading, not compiled, not run on hardware
//
#include <hip/hip_runtime.h>
#include <hip/hip_bf16.h>
#include <stdint.h>

// ---------------------------------------------------------------------------
// int8-weight linear: out[M,N] = (x[M,K]->bf16) @ (wq[N,K]->bf16)^T * scale[N]
// M=8192, N=4096, K=4096.  bf16 WMMA, f32 accumulate, per-channel scale in
// epilogue.  128x256 block tile, 8 wave32 (2x4), 64x64 per wave (4x4 WMMA
// frags), KT=32, double-buffered LDS software pipeline.
// Per wave per K-step: 16 v_wmma : 16 ds_load_b128 (1:1).
// ---------------------------------------------------------------------------

typedef __bf16 bf16;
typedef __attribute__((ext_vector_type(4)))  __bf16 v4bf;
typedef __attribute__((ext_vector_type(8)))  __bf16 v8bf;
typedef __attribute__((ext_vector_type(16))) __bf16 v16bf;
typedef __attribute__((ext_vector_type(8)))  float  v8f;

#define M_DIM 8192
#define N_DIM 4096
#define K_DIM 4096
#define BM 128
#define BN 256
#define KT 32
#define LDS_STRIDE 40            // KT + 8 bf16 pad (16B) -> bank stride 20
#define NK (K_DIM / KT)

__global__ __launch_bounds__(256) void qlinear_bf16_wmma(
    const float*  __restrict__ x,       // [M, K] fp32
    const int8_t* __restrict__ wq,      // [N, K] int8
    const float*  __restrict__ scaler,  // [N] fp32
    float*        __restrict__ out)     // [M, N] fp32
{
    __shared__ bf16 As[2][BM][LDS_STRIDE];   // 20 KB
    __shared__ bf16 Bs[2][BN][LDS_STRIDE];   // 40 KB

    const int tid   = threadIdx.x;
    const int lane  = tid & 31;
    const int wid   = tid >> 5;      // 0..7
    const int wm    = wid >> 2;      // 0..1  (64-row slab)
    const int wn    = wid & 3;       // 0..3  (64-col slab)
    const int lmod  = lane & 15;
    const int lhalf = lane >> 4;     // 0 or 1

    const int m0 = blockIdx.y * BM;
    const int n0 = blockIdx.x * BN;

    float4 areg[4];                  // 128x32 fp32 tile: 16 floats / thread
    int4   breg[2];                  // 256x32 int8 tile: 32 bytes / thread

    // -------- global fetch for tile kt ------------------------------------
    auto load_global = [&](int kt) {
        const int k0 = kt * KT;
#pragma unroll
        for (int i = 0; i < 4; ++i) {
            const int q   = tid + i * 256;    // 0..1023 float4 slots
            const int row = q >> 3;           // 0..127
            const int kq  = q & 7;            // 0..7 (x4 floats)
            areg[i] = *reinterpret_cast<const float4*>(
                x + (size_t)(m0 + row) * K_DIM + k0 + kq * 4);
        }
#pragma unroll
        for (int i = 0; i < 2; ++i) {
            const int q   = tid + i * 256;    // 0..511 int4 slots
            const int row = q >> 1;           // 0..255
            const int kh  = q & 1;            // 16 int8 per half
            breg[i] = *reinterpret_cast<const int4*>(
                wq + (size_t)(n0 + row) * K_DIM + k0 + kh * 16);
        }
    };

    // -------- convert + stage registers into LDS buffer `buf` --------------
    auto stage_lds = [&](int buf) {
#pragma unroll
        for (int i = 0; i < 4; ++i) {
            const int q   = tid + i * 256;
            const int row = q >> 3;
            const int kq  = q & 7;
            v4bf a4;
            a4.x = (bf16)areg[i].x;  a4.y = (bf16)areg[i].y;
            a4.z = (bf16)areg[i].z;  a4.w = (bf16)areg[i].w;
            *reinterpret_cast<v4bf*>(&As[buf][row][kq * 4]) = a4;
        }
#pragma unroll
        for (int i = 0; i < 2; ++i) {
            const int q   = tid + i * 256;
            const int row = q >> 1;
            const int kh  = q & 1;
            const int wv[4] = {breg[i].x, breg[i].y, breg[i].z, breg[i].w};
#pragma unroll
            for (int j = 0; j < 4; ++j) {
                const int v = wv[j];
                v4bf b4;                      // int8 -> bf16 is exact
                b4.x = (bf16)(float)(int8_t)(v);
                b4.y = (bf16)(float)(int8_t)(v >> 8);
                b4.z = (bf16)(float)(int8_t)(v >> 16);
                b4.w = (bf16)(float)(int8_t)(v >> 24);
                *reinterpret_cast<v4bf*>(&Bs[buf][row][kh * 16 + j * 4]) = b4;
            }
        }
    };

    const v8f zero8 = {0.f, 0.f, 0.f, 0.f, 0.f, 0.f, 0.f, 0.f};
    v8f acc[4][4];
#pragma unroll
    for (int mf = 0; mf < 4; ++mf)
#pragma unroll
        for (int nf = 0; nf < 4; ++nf)
            acc[mf][nf] = zero8;

    // ------------------------------- pipeline ------------------------------
    load_global(0);
    stage_lds(0);
    __syncthreads();

    for (int kt = 0; kt < NK; ++kt) {
        const int  cur  = kt & 1;
        const bool more = (kt + 1) < NK;
        if (more) load_global(kt + 1);        // overlap with WMMA below

        // A frag 16x32: lane<16 -> K{0..7,16..23}, lane>=16 -> K{8..15,24..31}
        v16bf af[4];
#pragma unroll
        for (int mf = 0; mf < 4; ++mf) {
            const bf16* pa = &As[cur][wm * 64 + mf * 16 + lmod][lhalf * 8];
            const v8bf lo = *reinterpret_cast<const v8bf*>(pa);
            const v8bf hi = *reinterpret_cast<const v8bf*>(pa + 16);
            af[mf] = __builtin_shufflevector(lo, hi,
                0, 1, 2, 3, 4, 5, 6, 7, 8, 9, 10, 11, 12, 13, 14, 15);
        }
        // B frag 32x16: lane = N, lanes 0-15 K{0..7,16..23}, lanes 16-31 K{8..15,24..31}
        v16bf bfr[4];
#pragma unroll
        for (int nf = 0; nf < 4; ++nf) {
            const bf16* pb = &Bs[cur][wn * 64 + nf * 16 + lmod][lhalf * 8];
            const v8bf lo = *reinterpret_cast<const v8bf*>(pb);
            const v8bf hi = *reinterpret_cast<const v8bf*>(pb + 16);
            bfr[nf] = __builtin_shufflevector(lo, hi,
                0, 1, 2, 3, 4, 5, 6, 7, 8, 9, 10, 11, 12, 13, 14, 15);
        }

#pragma unroll
        for (int mf = 0; mf < 4; ++mf)
#pragma unroll
            for (int nf = 0; nf < 4; ++nf)
                acc[mf][nf] = __builtin_amdgcn_wmma_f32_16x16x32_bf16(
                    false, af[mf], false, bfr[nf],
                    (short)0, acc[mf][nf], false, false);

        if (more) stage_lds(cur ^ 1);         // write opposite buffer
        __syncthreads();
    }

    // ------------------------ epilogue: scale + store ----------------------
    // C/D layout: lane 0-15 -> N=lane, M rows 0-7 in v[0..7];
    //             lane 16-31 -> N=lane-16, M rows 8-15.
#pragma unroll
    for (int nf = 0; nf < 4; ++nf) {
        const int   n_g = n0 + wn * 64 + nf * 16 + lmod;
        const float s   = scaler[n_g];
#pragma unroll
        for (int mf = 0; mf < 4; ++mf) {
            const int m_base = m0 + wm * 64 + mf * 16 + lhalf * 8;
            float* po = out + (size_t)m_base * N_DIM + n_g;
#pragma unroll
            for (int e = 0; e < 8; ++e)
                __builtin_nontemporal_store(acc[mf][nf][e] * s,
                                            po + (size_t)e * N_DIM);
        }
    }
}

// ---------------------------------------------------------------------------
extern "C" void kernel_launch(void* const* d_in, const int* in_sizes, int n_in,
                              void* d_out, int out_size, void* d_ws, size_t ws_size,
                              hipStream_t stream) {
    (void)in_sizes; (void)n_in; (void)out_size; (void)d_ws; (void)ws_size;
    const float*  x  = (const float*)d_in[0];
    const int8_t* wq = (const int8_t*)d_in[1];
    const float*  sc = (const float*)d_in[2];
    float* out = (float*)d_out;

    dim3 grid(N_DIM / BN, M_DIM / BM);   // 16 x 64 = 1024 workgroups
    qlinear_bf16_wmma<<<grid, 256, 0, stream>>>(x, wq, sc, out);
}
